// GraphEncoder_31147102830955
// MI455X (gfx1250) — compile-verified, matrix operand order checked by reading, not run on hardware
//
#include <hip/hip_runtime.h>
#include <hip/hip_bf16.h>

typedef __bf16 bf16;
typedef __attribute__((ext_vector_type(16))) __bf16 v16bf;
typedef __attribute__((ext_vector_type(8)))  float  v8f;
typedef __attribute__((ext_vector_type(4)))  int    i4;

#define AS1 __attribute__((address_space(1)))
#define AS3 __attribute__((address_space(3)))

#if __has_builtin(__builtin_amdgcn_global_load_async_to_lds_b128)
#define USE_ASYNC_LDS 1
#else
#define USE_ASYNC_LDS 0
#endif

union V16U { v16bf v; uint4 q[2]; };

__device__ __forceinline__ unsigned short f2bf(float x) {
    unsigned u = __float_as_uint(x);
    unsigned r = ((u >> 16) & 1u) + 0x7FFFu;   // round-to-nearest-even
    return (unsigned short)((u + r) >> 16);
}

__device__ __forceinline__ float fast_tanh(float x) {
#if __has_builtin(__builtin_amdgcn_tanhf)
    return __builtin_amdgcn_tanhf(x);          // gfx1250 v_tanh_f32
#elif __has_builtin(__builtin_amdgcn_tanh_f32)
    return __builtin_amdgcn_tanh_f32(x);
#else
    return tanhf(x);
#endif
}

__device__ __forceinline__ void wait_async0() {
#if __has_builtin(__builtin_amdgcn_s_wait_asynccnt)
    __builtin_amdgcn_s_wait_asynccnt(0);
#else
    asm volatile("s_wait_asynccnt 0x0" ::: "memory");
#endif
}

// ---------------- elementwise helpers ----------------

__global__ void k_zero_f32(float* p, long long n) {
    long long i = (long long)blockIdx.x * blockDim.x + threadIdx.x;
    if (i < n) p[i] = 0.f;
}

// weight convert + transpose: w[K,N] (f32 row-major) -> wt[N,K] (bf16 row-major)
__global__ void k_w_to_bf16_t(const float* __restrict__ w, unsigned short* __restrict__ wt,
                              int K, int N) {
    long long i = (long long)blockIdx.x * blockDim.x + threadIdx.x;
    if (i >= (long long)K * N) return;
    int k = (int)(i / N), n = (int)(i % N);
    wt[(long long)n * K + k] = f2bf(w[i]);
}

// agg[dst] += x[src] over all edges (coalesced over feature dim); C = 1<<cshift
__global__ void k_edge_scatter(const float* __restrict__ x, const long long* __restrict__ ei,
                               float* __restrict__ agg, int E, int cshift) {
    long long i = (long long)blockIdx.x * blockDim.x + threadIdx.x;
    long long tot = (long long)E << cshift;
    if (i >= tot) return;
    int e = (int)(i >> cshift);
    int j = (int)(i & ((1 << cshift) - 1));
    long long s = ei[e];
    long long d = ei[(long long)E + e];
    atomicAdd(&agg[(d << cshift) + j], x[(s << cshift) + j]);
}

// h = x + agg  (bf16 output, becomes WMMA A operand)
__global__ void k_add_to_bf16(const float* __restrict__ x, const float* __restrict__ agg,
                              unsigned short* __restrict__ h, long long n) {
    long long i = (long long)blockIdx.x * blockDim.x + threadIdx.x;
    if (i < n) h[i] = f2bf(x[i] + agg[i]);
}

// ---------------- BatchNorm (training-mode, biased var) ----------------

__global__ void k_bn_partial(const float* __restrict__ z, float* __restrict__ sums,
                             float* __restrict__ sqs, int M, int C) {
    int j = threadIdx.x;              // C == blockDim.x == 256
    int r0 = blockIdx.x * 256;
    int r1 = r0 + 256; if (r1 > M) r1 = M;
    float s = 0.f, q = 0.f;
    for (int r = r0; r < r1; ++r) {
        float v = z[(long long)r * C + j];
        s += v; q += v * v;
    }
    atomicAdd(&sums[j], s);
    atomicAdd(&sqs[j],  q);
}

__global__ void k_bn_final(const float* __restrict__ sums, const float* __restrict__ sqs,
                           float* __restrict__ mu, float* __restrict__ inv, int M) {
    int j = threadIdx.x;
    float m = sums[j] / (float)M;
    float v = sqs[j] / (float)M - m * m;
    mu[j]  = m;
    inv[j] = rsqrtf(v + 1e-5f);
}

// out_bf16 = relu((z - mu) * inv * g + be), C = 256
__global__ void k_bn_apply(const float* __restrict__ z, const float* __restrict__ mu,
                           const float* __restrict__ inv, const float* __restrict__ g,
                           const float* __restrict__ be, unsigned short* __restrict__ out,
                           long long n) {
    long long i = (long long)blockIdx.x * blockDim.x + threadIdx.x;
    if (i >= n) return;
    int j = (int)(i & 255);
    float v = (z[i] - mu[j]) * inv[j] * g[j] + be[j];
    out[i] = f2bf(fmaxf(v, 0.f));
}

// ---------------- pooling (C = 256) ----------------

__global__ void k_pool_scatter(const float* __restrict__ act, const long long* __restrict__ batch,
                               float* __restrict__ psum, float* __restrict__ cnt, int N) {
    long long i = (long long)blockIdx.x * blockDim.x + threadIdx.x;
    long long tot = (long long)N << 8;
    if (i >= tot) return;
    int node = (int)(i >> 8), j = (int)(i & 255);
    long long b = batch[node];
    atomicAdd(&psum[(b << 8) + j], act[i]);
    if (j == 0) atomicAdd(&cnt[b], 1.f);
}

__global__ void k_pool_final(const float* __restrict__ psum, const float* __restrict__ cnt,
                             unsigned short* __restrict__ out, int G) {
    long long i = (long long)blockIdx.x * blockDim.x + threadIdx.x;
    if (i >= ((long long)G << 8)) return;
    float c = cnt[i >> 8];
    out[i] = f2bf(psum[i] / fmaxf(c, 1.f));
}

// ---------------- WMMA GEMM ----------------
// C[M,N] = act( A[M,K](bf16 row-major) x B^T[N,K](bf16 row-major) + bias )
// Block: 256 threads = 8 wave32. Macro-tile 128(M) x 32(N); each wave owns a
// 16x32 strip = 2 accumulators (A operand reused across two B operands).
// Staging: async global->LDS (ASYNCcnt) when available; operands read as
// contiguous ds_load_b128 pairs per the CDNA5 16-bit WMMA layout.
// act: 0=none 1=relu 2=tanh 3=relu(tanh)
#define LDSTRIDE 40   // bf16 elems: 80B rows -> conflict-free b128 reads, 16B aligned
__launch_bounds__(256)
__global__ void k_gemm_wmma(const bf16* __restrict__ A, const bf16* __restrict__ Bt,
                            const float* __restrict__ bias,
                            float* __restrict__ Cf, unsigned short* __restrict__ Cbf,
                            int M, int N, int K, int act) {
    __shared__ __align__(16) bf16 As[128 * LDSTRIDE];  // 128 rows x 32 K
    __shared__ __align__(16) bf16 Bs[32 * LDSTRIDE];   // 32 out-cols x 32 K

    const int tid  = threadIdx.x;
    const int wave = tid >> 5;
    const int lane = tid & 31;
    const int l    = lane & 15;
    const int hi   = lane >> 4;
    const int kb   = hi * 8;          // lanes 0-15: K0..7|K16..23 ; 16-31: K8..15|K24..31
    const int m0   = blockIdx.x * 128;
    const int n0   = blockIdx.y * 32;

    v8f acc0 = {}, acc1 = {};

    for (int k0 = 0; k0 < K; k0 += 32) {
#if USE_ASYNC_LDS
        // async staging: DMA straight into LDS, no VGPR round-trip.
        // Tail rows (>= M) stay stale; their garbage only reaches masked D rows.
        #pragma unroll
        for (int c = tid; c < 512; c += 256) {
            int row = c >> 2, q = c & 3;
            int grow = m0 + row;
            if (grow < M)
                __builtin_amdgcn_global_load_async_to_lds_b128(
                    (AS1 i4*)(A + (long long)grow * K + k0 + q * 8),
                    (AS3 i4*)&As[row * LDSTRIDE + q * 8], 0, 0);
        }
        if (tid < 128) {
            int row = tid >> 2, q = tid & 3;
            __builtin_amdgcn_global_load_async_to_lds_b128(
                (AS1 i4*)(Bt + (long long)(n0 + row) * K + k0 + q * 8),
                (AS3 i4*)&Bs[row * LDSTRIDE + q * 8], 0, 0);
        }
        wait_async0();
#else
        #pragma unroll
        for (int c = tid; c < 512; c += 256) {
            int row = c >> 2, q = c & 3;
            uint4 av = {0u, 0u, 0u, 0u};
            int grow = m0 + row;
            if (grow < M)
                av = *(const uint4*)(A + (long long)grow * K + k0 + q * 8);
            *(uint4*)&As[row * LDSTRIDE + q * 8] = av;
        }
        if (tid < 128) {
            int row = tid >> 2, q = tid & 3;
            uint4 bv = *(const uint4*)(Bt + (long long)(n0 + row) * K + k0 + q * 8);
            *(uint4*)&Bs[row * LDSTRIDE + q * 8] = bv;
        }
#endif
        __syncthreads();

        const bf16* arow = &As[(wave * 16 + l) * LDSTRIDE];
        const bf16* br0  = &Bs[l * LDSTRIDE];
        const bf16* br1  = &Bs[(l + 16) * LDSTRIDE];
        V16U a, b0, b1;
        a.q[0]  = *(const uint4*)&arow[kb]; a.q[1]  = *(const uint4*)&arow[kb + 16];
        b0.q[0] = *(const uint4*)&br0[kb];  b0.q[1] = *(const uint4*)&br0[kb + 16];
        b1.q[0] = *(const uint4*)&br1[kb];  b1.q[1] = *(const uint4*)&br1[kb + 16];

        acc0 = __builtin_amdgcn_wmma_f32_16x16x32_bf16(false, a.v, false, b0.v, (short)0, acc0, false, false);
        acc1 = __builtin_amdgcn_wmma_f32_16x16x32_bf16(false, a.v, false, b1.v, (short)0, acc1, false, false);
        __syncthreads();
    }

    const int col0 = n0 + l, col1 = n0 + 16 + l;
    const float bv0 = bias ? bias[col0] : 0.f;
    const float bv1 = bias ? bias[col1] : 0.f;
    #pragma unroll
    for (int v = 0; v < 8; ++v) {
        int r = m0 + wave * 16 + hi * 8 + v;   // C/D layout: VGPR v -> row v (+8 upper half)
        if (r < M) {
            float x0 = acc0[v] + bv0;
            float x1 = acc1[v] + bv1;
            if      (act == 1) { x0 = fmaxf(x0, 0.f);            x1 = fmaxf(x1, 0.f); }
            else if (act == 2) { x0 = fast_tanh(x0);             x1 = fast_tanh(x1); }
            else if (act == 3) { x0 = fmaxf(fast_tanh(x0), 0.f); x1 = fmaxf(fast_tanh(x1), 0.f); }
            if (Cf) {
                Cf[(long long)r * N + col0] = x0;
                Cf[(long long)r * N + col1] = x1;
            }
            if (Cbf) {
                Cbf[(long long)r * N + col0] = f2bf(x0);
                Cbf[(long long)r * N + col1] = f2bf(x1);
            }
        }
    }
}

// ---------------- host launcher ----------------

extern "C" void kernel_launch(void* const* d_in, const int* in_sizes, int n_in,
                              void* d_out, int out_size, void* d_ws, size_t ws_size,
                              hipStream_t stream) {
    (void)n_in; (void)out_size; (void)ws_size;
    const int F = 128, C = 256, H = 512, O = 768, G = 512;
    const int N = in_sizes[0] / F;
    const int E = in_sizes[1] / 2;

    const float*     x     = (const float*)d_in[0];
    const long long* ei    = (const long long*)d_in[1];
    const long long* batch = (const long long*)d_in[2];
    const float *w1a=(const float*)d_in[3],  *b1a=(const float*)d_in[4];
    const float *g1 =(const float*)d_in[5],  *be1=(const float*)d_in[6];
    const float *w1b=(const float*)d_in[7],  *b1b=(const float*)d_in[8];
    const float *w2a=(const float*)d_in[9],  *b2a=(const float*)d_in[10];
    const float *g2 =(const float*)d_in[11], *be2=(const float*)d_in[12];
    const float *w2b=(const float*)d_in[13], *b2b=(const float*)d_in[14];
    const float *w3a=(const float*)d_in[15], *b3a=(const float*)d_in[16];
    const float *g3 =(const float*)d_in[17], *be3=(const float*)d_in[18];
    const float *w3b=(const float*)d_in[19], *b3b=(const float*)d_in[20];
    const float *wh1=(const float*)d_in[21], *bh1=(const float*)d_in[22];
    const float *wh2=(const float*)d_in[23], *bh2=(const float*)d_in[24];

    char* ws = (char*)d_ws;
    size_t off = 0;
    auto alloc = [&](size_t bytes) -> void* {
        void* p = ws + off;
        off += (bytes + 255) & ~(size_t)255;
        return p;
    };

    // transposed bf16 weights (B^T, N x K row-major)
    unsigned short* w1a_bf = (unsigned short*)alloc((size_t)F * C * 2);
    unsigned short* w1b_bf = (unsigned short*)alloc((size_t)C * C * 2);
    unsigned short* w2a_bf = (unsigned short*)alloc((size_t)C * C * 2);
    unsigned short* w2b_bf = (unsigned short*)alloc((size_t)C * C * 2);
    unsigned short* w3a_bf = (unsigned short*)alloc((size_t)C * C * 2);
    unsigned short* w3b_bf = (unsigned short*)alloc((size_t)C * C * 2);
    unsigned short* wh1_bf = (unsigned short*)alloc((size_t)C * H * 2);
    unsigned short* wh2_bf = (unsigned short*)alloc((size_t)H * O * 2);

    float*          buf1   = (float*)alloc((size_t)N * C * 4);          // agg, then GEMM-1 out
    unsigned short* hbf    = (unsigned short*)alloc((size_t)N * C * 2); // h bf16, then BN-out bf16
    float*          act    = (float*)alloc((size_t)N * C * 4);          // layer activations
    float*          stat   = (float*)alloc((size_t)4 * C * 4);
    float* sums = stat, * sqs = stat + C, * mu = stat + 2 * C, * inv = stat + 3 * C;
    float*          psum   = (float*)alloc((size_t)G * C * 4);
    float*          cnt    = (float*)alloc((size_t)G * 4);
    unsigned short* pooled = (unsigned short*)alloc((size_t)G * C * 2);
    unsigned short* h1_bf  = (unsigned short*)alloc((size_t)G * H * 2);

    auto blocks = [](long long n) { return (unsigned)((n + 255) / 256); };
    auto cvt = [&](const float* src, unsigned short* dst, int K, int Nn) {
        k_w_to_bf16_t<<<blocks((long long)K * Nn), 256, 0, stream>>>(src, dst, K, Nn);
    };
    cvt(w1a, w1a_bf, F, C);
    cvt(w1b, w1b_bf, C, C);
    cvt(w2a, w2a_bf, C, C);
    cvt(w2b, w2b_bf, C, C);
    cvt(w3a, w3a_bf, C, C);
    cvt(w3b, w3b_bf, C, C);
    cvt(wh1, wh1_bf, C, H);
    cvt(wh2, wh2_bf, H, O);

    auto gin_layer = [&](const float* in, int Fin, int fshift,
                         const unsigned short* wa, const float* ba,
                         const float* g, const float* be,
                         const unsigned short* wb, const float* bb, int actMode) {
        long long nF = (long long)N * Fin;
        long long eF = (long long)E * Fin;
        long long nC = (long long)N * C;
        k_zero_f32<<<blocks(nF), 256, 0, stream>>>(buf1, nF);
        k_edge_scatter<<<blocks(eF), 256, 0, stream>>>(in, ei, buf1, E, fshift);
        k_add_to_bf16<<<blocks(nF), 256, 0, stream>>>(in, buf1, hbf, nF);
        k_gemm_wmma<<<dim3((N + 127) / 128, C / 32), 256, 0, stream>>>(
            (const bf16*)hbf, (const bf16*)wa, ba, buf1, nullptr, N, C, Fin, 0);
        k_zero_f32<<<2, 256, 0, stream>>>(sums, 2 * C);
        k_bn_partial<<<(N + 255) / 256, 256, 0, stream>>>(buf1, sums, sqs, N, C);
        k_bn_final<<<1, C, 0, stream>>>(sums, sqs, mu, inv, N);
        k_bn_apply<<<blocks(nC), 256, 0, stream>>>(buf1, mu, inv, g, be, hbf, nC);
        k_gemm_wmma<<<dim3((N + 127) / 128, C / 32), 256, 0, stream>>>(
            (const bf16*)hbf, (const bf16*)wb, bb, act, nullptr, N, C, C, actMode);
    };

    gin_layer(x,   F, 7, w1a_bf, b1a, g1, be1, w1b_bf, b1b, 1); // relu
    gin_layer(act, C, 8, w2a_bf, b2a, g2, be2, w2b_bf, b2b, 3); // relu(tanh)
    gin_layer(act, C, 8, w3a_bf, b3a, g3, be3, w3b_bf, b3b, 2); // tanh

    // global mean pool
    long long gc = (long long)G * C;
    long long nc = (long long)N * C;
    k_zero_f32<<<blocks(gc), 256, 0, stream>>>(psum, gc);
    k_zero_f32<<<blocks(G), 256, 0, stream>>>(cnt, G);
    k_pool_scatter<<<blocks(nc), 256, 0, stream>>>(act, batch, psum, cnt, N);
    k_pool_final<<<blocks(gc), 256, 0, stream>>>(psum, cnt, pooled, G);

    // head: relu(pooled @ wh1 + bh1) @ wh2 + bh2
    k_gemm_wmma<<<dim3((G + 127) / 128, H / 32), 256, 0, stream>>>(
        (const bf16*)pooled, (const bf16*)wh1_bf, bh1, nullptr, h1_bf, G, H, C, 1);
    k_gemm_wmma<<<dim3((G + 127) / 128, O / 32), 256, 0, stream>>>(
        (const bf16*)h1_bf, (const bf16*)wh2_bf, bh2, (float*)d_out, nullptr, G, O, H, 0);
}
